// MOE_31413390803110
// MI455X (gfx1250) — matrix operand order repaired and checked by reading, not compiled, not run on hardware
//
#include <hip/hip_runtime.h>
#include <hip/hip_bf16.h>

#define BT_ 8192   // B*T = 4*2048
#define C_  1024
#define H_  2048
#define E_  8
#define WLD 40     // padded LDS row stride (bf16) for B tiles: 80B, 16B-aligned

typedef __attribute__((ext_vector_type(16))) __bf16 v16bf;
typedef __attribute__((ext_vector_type(8)))  float  v8f;

// ---------------- workspace layout (bytes) ----------------
#define OFF_WE   (0)                         // BT*E f32  = 256 KB
#define OFF_CNT  (256*1024)                  // E ints (4 KB reserved)
#define OFF_IDX  (260*1024)                  // E*BT ints = 256 KB
#define OFF_XB   (516*1024)                  // BT*C bf16 = 16 MB
#define OFF_WFC  (OFF_XB + (size_t)BT_*C_*2)         // E*H*C bf16 = 32 MB
#define OFF_WPJ  (OFF_WFC + (size_t)E_*C_*H_*2)      // E*C*H bf16 = 32 MB

// -------- CDNA5 async global->LDS copy (16B per lane) + wait --------
__device__ __forceinline__ void async_copy16(void* ldsDst, const void* gSrc) {
  unsigned lds = (unsigned)(unsigned long long)(uintptr_t)ldsDst;  // LDS byte offset = low 32 bits of flat addr
  asm volatile("global_load_async_to_lds_b128 %0, %1, off"
               :: "v"(lds), "v"(gSrc) : "memory");
}
__device__ __forceinline__ void wait_async0() {
  asm volatile("s_wait_asynccnt 0x0" ::: "memory");
}

// fast silu: v * rcp(1+exp(-v)) — single v_rcp_f32, plenty for bf16 output
__device__ __forceinline__ float fast_silu(float v) {
  return v * __builtin_amdgcn_rcpf(1.f + __expf(-v));
}

// ---------------- converters: f32 -> bf16, expert-major, PRE-TRANSPOSED ----------------
__global__ void cvt_x_kernel(const float* __restrict__ x, __bf16* __restrict__ xb) {
  size_t i = (size_t)blockIdx.x * blockDim.x + threadIdx.x;   // over BT*C/4
  if (i >= (size_t)BT_ * C_ / 4) return;
  const float4 f = *(const float4*)(x + 4 * i);
  __bf16* o = xb + 4 * i;
  o[0] = (__bf16)f.x; o[1] = (__bf16)f.y; o[2] = (__bf16)f.z; o[3] = (__bf16)f.w;
}

// w_fc (C,H,E) f32 -> wfcb[e][h][c] bf16   (transposed: c contiguous)
__global__ void cvt_wfc_kernel(const float* __restrict__ w, __bf16* __restrict__ o) {
  size_t i = (size_t)blockIdx.x * blockDim.x + threadIdx.x;   // out linear over (e,h,c)
  if (i >= (size_t)E_ * H_ * C_) return;
  int c = (int)(i % C_);
  size_t t = i / C_;
  int h = (int)(t % H_);
  int e = (int)(t / H_);
  o[i] = (__bf16)w[((size_t)c * H_ + h) * E_ + e];
}

// w_proj (H,C,E) f32 -> wpjb[e][c][h] bf16  (transposed: h contiguous)
__global__ void cvt_wpj_kernel(const float* __restrict__ w, __bf16* __restrict__ o) {
  size_t i = (size_t)blockIdx.x * blockDim.x + threadIdx.x;   // out linear over (e,c,h)
  if (i >= (size_t)E_ * C_ * H_) return;
  int h = (int)(i % H_);
  size_t t = i / H_;
  int c = (int)(t % C_);
  int e = (int)(t / C_);
  o[i] = (__bf16)w[((size_t)h * C_ + c) * E_ + e];
}

// ---------------- gating: logits, top-2 (raw logits as weights), token lists ----------------
__global__ void gate_kernel(const float* __restrict__ x, const float* __restrict__ wg,
                            float* __restrict__ wE, int* __restrict__ cnt,
                            int* __restrict__ idxArr) {
  int tok = blockIdx.x * blockDim.x + threadIdx.x;
  if (tok >= BT_) return;
  float g[E_];
#pragma unroll
  for (int e = 0; e < E_; ++e) g[e] = 0.f;
  const float* xr = x + (size_t)tok * C_;
  for (int c = 0; c < C_; ++c) {
    float xv = xr[c];
    const float* wr = wg + (size_t)c * E_;
#pragma unroll
    for (int e = 0; e < E_; ++e) g[e] += xv * wr[e];
  }
  int i1 = 0; float v1 = g[0];
#pragma unroll
  for (int e = 1; e < E_; ++e) if (g[e] > v1) { v1 = g[e]; i1 = e; }
  int i2 = -1; float v2 = -3.4e38f;
#pragma unroll
  for (int e = 0; e < E_; ++e) if (e != i1 && g[e] > v2) { v2 = g[e]; i2 = e; }
  float* we = wE + (size_t)tok * E_;
#pragma unroll
  for (int e = 0; e < E_; ++e) we[e] = 0.f;
  we[i1] = v1;
  we[i2] = v2;
  int p1 = atomicAdd(&cnt[i1], 1);
  idxArr[(size_t)i1 * BT_ + p1] = tok;
  int p2 = atomicAdd(&cnt[i2], 1);
  idxArr[(size_t)i2 * BT_ + p2] = tok;
}

// ---------------- WMMA fragment loaders (CDNA5 16-bit layouts) ----------------
// A: 16(M)x32(K) tile, row-major in LDS with row stride ld.
// Lane L: m = L%16, half = L/16; VGPR v holds K = (v/4)*16 + half*8 + 2*(v%4), +1.
__device__ __forceinline__ v16bf fragA(const __bf16* p, int ld, int lane) {
  int m = lane & 15, half = lane >> 4;
  const __bf16* row = p + m * ld;
  v16bf a;
#pragma unroll
  for (int v = 0; v < 8; ++v) {
    int k = ((v >> 2) << 4) + (half << 3) + ((v & 3) << 1);
    a[2 * v]     = row[k];
    a[2 * v + 1] = row[k + 1];
  }
  return a;
}

// B: 32(K)x16(N) tile staged TRANSPOSED in LDS as [N][K] with row stride ld.
// Lane L: n = L%16, half = L/16; VGPR v holds K = (v/4)*16 + half*8 + 2*(v%4), +1.
__device__ __forceinline__ v16bf fragB(const __bf16* pT, int ld, int lane) {
  int n = lane & 15, half = lane >> 4;
  const __bf16* col = pT + n * ld;
  v16bf b;
#pragma unroll
  for (int v = 0; v < 8; ++v) {
    int k = ((v >> 2) << 4) + (half << 3) + ((v & 3) << 1);
    b[2 * v]     = col[k];
    b[2 * v + 1] = col[k + 1];
  }
  return b;
}

// ---------------- fused expert kernel ----------------
// grid: (64 token-tiles, H/512 chunks, E experts); block: 256 threads (8 wave32).
// Phase 1: h = silu(Xg(128x1024) @ Wfc[:, hchunk(512)]) * w_tok   -> LDS (bf16)
// Phase 2: out += h(128x512) @ Wproj[hchunk, :](512x1024)         -> f32 atomics
__global__ __launch_bounds__(256) void moe_expert_kernel(
    const __bf16* __restrict__ xb, const __bf16* __restrict__ wfc,
    const __bf16* __restrict__ wpj, const float* __restrict__ wE,
    const int* __restrict__ cnt, const int* __restrict__ idxArr,
    float* __restrict__ out) {
  __shared__ int   s_idx[128];
  __shared__ float s_w[128];
  __shared__ __align__(16) __bf16 xs[128 * 32];      //  8 KB : A staging
  __shared__ __align__(16) __bf16 wts[256 * WLD];    // 20 KB : B staging ([n][k], padded)
  __shared__ __align__(16) __bf16 hbuf[128 * 512];   // 128 KB : silu(h)*w, bf16

  const int e     = blockIdx.z;
  const int tile  = blockIdx.x;
  const int hbase = blockIdx.y * 512;
  const int n_e   = cnt[e];
  if (tile * 128 >= n_e) return;  // uniform early-exit (before any barrier)

  const int tid  = threadIdx.x;
  const int lane = tid & 31;
  const int wv   = tid >> 5;
  const int wM   = wv >> 2, wN = wv & 3;       // 2x4 wave grid, each wave 64x64
  const int half = lane >> 4, n16 = lane & 15;

  if (tid < 128) {
    int g = tile * 128 + tid;
    if (g < n_e) {
      int t = idxArr[(size_t)e * BT_ + g];
      s_idx[tid] = t;
      s_w[tid]   = wE[(size_t)t * E_ + e];
    } else {
      s_idx[tid] = 0;      // pad rows: weight 0 => h row exactly 0 => contributes 0
      s_w[tid]   = 0.f;
    }
  }
  __syncthreads();

  // Hoisted per-thread staging addresses (loop-invariant).
  const int r0 = tid >> 2, c8 = (tid & 3) << 3;          // X: 2 rows/thread, 16B each
  const __bf16* xrow0 = xb + (size_t)s_idx[r0]      * C_ + c8;
  const __bf16* xrow1 = xb + (size_t)s_idx[r0 + 64] * C_ + c8;
  __bf16* xd0 = xs + r0 * 32 + c8;
  __bf16* xd1 = xs + (r0 + 64) * 32 + c8;

  // ---------------- Phase 1: FC GEMM + silu + scale -> hbuf ----------------
  for (int hblk = 0; hblk < 2; ++hblk) {      // two 256-wide sub-blocks of the 512 chunk
    v8f acc[4][4];
    const v8f vzero = {0.f, 0.f, 0.f, 0.f, 0.f, 0.f, 0.f, 0.f};
#pragma unroll
    for (int i = 0; i < 4; ++i)
#pragma unroll
      for (int j = 0; j < 4; ++j) acc[i][j] = vzero;

    // B source: wfcb[e][h][c], rows n = h in [colbase, colbase+256), cols k = c
    const __bf16* srcW = wfc + ((size_t)e * H_ + hbase + hblk * 256) * C_;
    for (int kk = 0; kk < C_; kk += 32) {
      __syncthreads();
      // async stage A: gathered token rows, 128x32 bf16 (2 x 16B per thread)
      async_copy16(xd0, xrow0 + kk);
      async_copy16(xd1, xrow1 + kk);
      // async stage B: 256 rows x 32 cols (contiguous 64B rows, 4 x 16B per thread)
      {
        int n = tid >> 2, kc = (tid & 3) << 3;
#pragma unroll
        for (int rr = 0; rr < 4; ++rr)                   // rows n, n+64, n+128, n+192
          async_copy16(wts + (n + rr * 64) * WLD + kc,
                       srcW + (size_t)(n + rr * 64) * C_ + kk + kc);
      }
      wait_async0();
      __syncthreads();

      v16bf af[4], bf[4];
#pragma unroll
      for (int i = 0; i < 4; ++i) af[i] = fragA(xs + (wM * 64 + i * 16) * 32, 32, lane);
#pragma unroll
      for (int j = 0; j < 4; ++j) bf[j] = fragB(wts + (wN * 64 + j * 16) * WLD, WLD, lane);
#pragma unroll
      for (int i = 0; i < 4; ++i)
#pragma unroll
        for (int j = 0; j < 4; ++j)
          acc[i][j] = __builtin_amdgcn_wmma_f32_16x16x32_bf16(
              false, af[i], false, bf[j], (short)0, acc[i][j], false, false);
    }
    __syncthreads();

    // silu * token-weight -> hbuf (bf16), fast rcp-based sigmoid
#pragma unroll
    for (int i = 0; i < 4; ++i) {
      const int mBase = wM * 64 + i * 16 + half * 8;
#pragma unroll
      for (int j = 0; j < 4; ++j) {
        const int col = hblk * 256 + wN * 64 + j * 16 + n16;
#pragma unroll
        for (int r = 0; r < 8; ++r) {
          const int m = mBase + r;
          float s = fast_silu(acc[i][j][r]) * s_w[m];
          hbuf[m * 512 + col] = (__bf16)s;
        }
      }
    }
  }
  __syncthreads();

  // ---------------- Phase 2: proj GEMM, atomic accumulate ----------------
  for (int nb = 0; nb < 4; ++nb) {            // 4 x 256-wide output blocks over C
    v8f acc[4][4];
    const v8f vzero = {0.f, 0.f, 0.f, 0.f, 0.f, 0.f, 0.f, 0.f};
#pragma unroll
    for (int i = 0; i < 4; ++i)
#pragma unroll
      for (int j = 0; j < 4; ++j) acc[i][j] = vzero;

    // B source: wpjb[e][c][h], rows n = c in [nb*256, +256), cols k = h (hbase+kk..)
    const __bf16* srcW = wpj + ((size_t)e * C_ + nb * 256) * H_ + hbase;
    for (int kk = 0; kk < 512; kk += 32) {
      __syncthreads();
      {
        int n = tid >> 2, kc = (tid & 3) << 3;
#pragma unroll
        for (int rr = 0; rr < 4; ++rr)
          async_copy16(wts + (n + rr * 64) * WLD + kc,
                       srcW + (size_t)(n + rr * 64) * H_ + kk + kc);
      }
      wait_async0();
      __syncthreads();

      v16bf af[4], bf[4];
#pragma unroll
      for (int i = 0; i < 4; ++i)
        af[i] = fragA(hbuf + (wM * 64 + i * 16) * 512 + kk, 512, lane);  // A direct from hbuf
#pragma unroll
      for (int j = 0; j < 4; ++j) bf[j] = fragB(wts + (wN * 64 + j * 16) * WLD, WLD, lane);
#pragma unroll
      for (int i = 0; i < 4; ++i)
#pragma unroll
        for (int j = 0; j < 4; ++j)
          acc[i][j] = __builtin_amdgcn_wmma_f32_16x16x32_bf16(
              false, af[i], false, bf[j], (short)0, acc[i][j], false, false);
    }

    // scatter-accumulate to out rows (f32 atomics)
#pragma unroll
    for (int i = 0; i < 4; ++i) {
      const int mBase = wM * 64 + i * 16 + half * 8;
#pragma unroll
      for (int j = 0; j < 4; ++j) {
        const int col = nb * 256 + wN * 64 + j * 16 + n16;
#pragma unroll
        for (int r = 0; r < 8; ++r) {
          const int m   = mBase + r;
          const int tok = s_idx[m];
          __hip_atomic_fetch_add(&out[(size_t)tok * C_ + col], acc[i][j][r],
                                 __ATOMIC_RELAXED, __HIP_MEMORY_SCOPE_AGENT);
        }
      }
    }
  }
}

// ---------------- launcher ----------------
extern "C" void kernel_launch(void* const* d_in, const int* in_sizes, int n_in,
                              void* d_out, int out_size, void* d_ws, size_t ws_size,
                              hipStream_t stream) {
  const float* x      = (const float*)d_in[0];  // (B,T,C)
  const float* w_gate = (const float*)d_in[1];  // (C,E)
  const float* w_fc   = (const float*)d_in[2];  // (C,H,E)
  const float* w_proj = (const float*)d_in[3];  // (H,C,E)
  float* out = (float*)d_out;                   // (B,T,C) f32

  char* ws = (char*)d_ws;
  float*  wE   = (float*)(ws + OFF_WE);
  int*    cnt  = (int*)(ws + OFF_CNT);
  int*    idxA = (int*)(ws + OFF_IDX);
  __bf16* xb   = (__bf16*)(ws + OFF_XB);
  __bf16* wfcb = (__bf16*)(ws + OFF_WFC);
  __bf16* wpjb = (__bf16*)(ws + OFF_WPJ);

  hipMemsetAsync(out, 0, (size_t)out_size * sizeof(float), stream);
  hipMemsetAsync(cnt, 0, 4096, stream);

  cvt_x_kernel<<<(BT_ * C_ / 4 + 255) / 256, 256, 0, stream>>>(x, xb);
  cvt_wfc_kernel<<<(E_ * H_ * C_ + 255) / 256, 256, 0, stream>>>(w_fc, wfcb);
  cvt_wpj_kernel<<<(E_ * C_ * H_ + 255) / 256, 256, 0, stream>>>(w_proj, wpjb);
  gate_kernel<<<BT_ / 256, 256, 0, stream>>>(x, w_gate, wE, cnt, idxA);

  moe_expert_kernel<<<dim3(64, 4, 8), 256, 0, stream>>>(xb, wfcb, wpjb, wE, cnt, idxA, out);
}
